// Decoder_43035572306781
// MI455X (gfx1250) — compile-verified, matrix operand order checked by reading, not run on hardware
//
#include <hip/hip_runtime.h>
#include <hip/hip_bf16.h>
#include <stdint.h>

typedef __attribute__((ext_vector_type(16))) __bf16 v16bf;
typedef __attribute__((ext_vector_type(8)))  float  v8f;
typedef __attribute__((ext_vector_type(4)))  unsigned u32x4;
typedef __attribute__((ext_vector_type(4)))  int      i32x4;
typedef __attribute__((ext_vector_type(8)))  int      i32x8;

#define Bq 128
#define Sq 512
#define Uq 512
#define Vq 64
#define NWG 104   // 32 per LSTM layer (x3) + 8 projection
#define TPB 128

// ---------- helpers ----------
__device__ __forceinline__ unsigned short f2bf(float f) {
  union { float f; unsigned u; } x; x.f = f;
  unsigned u = x.u;
  return (unsigned short)((u + 0x7FFFu + ((u >> 16) & 1u)) >> 16);
}
__device__ __forceinline__ float fsig(float x) { return 1.f / (1.f + __expf(-x)); }
__device__ __forceinline__ float ftanh(float x) {
  float ax = fabsf(x);
  float e = __expf(-2.f * ax);
  float r = (1.f - e) / (1.f + e);
  return x < 0.f ? -r : r;
}

union FragU { v16bf v; uint4 q[2]; };

// A fragment from LDS tile (rows packed at rstride elems). Emits ds_load_b128.
__device__ __forceinline__ v16bf load_a_lds(const unsigned short* s, int rstride,
                                            int bb, int k0, int lane) {
  int m = lane & 15, hk = lane >> 4;
  const unsigned short* p = s + (bb * 16 + m) * rstride + k0 + hk * 8;
  FragU f;
  f.q[0] = *(const uint4*)(p);
  f.q[1] = *(const uint4*)(p + 16);
  return f.v;
}
// B fragment: weights pre-packed so each lane's 16 bf16 are contiguous.
__device__ __forceinline__ v16bf load_b_frag(const unsigned short* pack, int kt,
                                             int nt, int ntiles, int lane) {
  const unsigned short* p = pack + (((size_t)kt * ntiles + nt) << 9) + lane * 16;
  FragU f;
  f.q[0] = *(const uint4*)(p);
  f.q[1] = *(const uint4*)(p + 8);
  return f.v;
}
__device__ __forceinline__ v8f wmma_bf16(v16bf a, v16bf b, v8f c) {
  return __builtin_amdgcn_wmma_f32_16x16x32_bf16(false, a, false, b, (short)0, c,
                                                 false, false);
}

__device__ __forceinline__ unsigned ldsoff(const void* p) {
  // generic LDS pointers carry the byte offset in the low 32 bits
  return (unsigned)(uintptr_t)p;
}

// ---- Tensor Data Mover: 2D tile (rows x tile_k bf16) global -> LDS ----
// D# built per CDNA5 ISA 8.3/8.4 (group0: count/lds/global/type, group1:
// data_size=2B, tensor dims, tile dims, dim0 stride). Groups 2/3 zero (2D).
__device__ __forceinline__ void tdm_load_2d(unsigned lds_byte_off, const void* g,
                                            unsigned tile_k, unsigned rows,
                                            unsigned long long stride_elems) {
  unsigned long long ga = (unsigned long long)(uintptr_t)g;
  u32x4 g0;
  g0.x = 1u;                                   // count=1 valid descriptor
  g0.y = lds_byte_off;                         // lds_addr
  g0.z = (unsigned)ga;                         // global_addr[31:0]
  g0.w = (unsigned)((ga >> 32) & 0x01FFFFFFu) | (2u << 30);  // addr[56:32]|type=2
  unsigned dim0 = (unsigned)stride_elems;      // tensor width (elements)
  unsigned dim1 = rows;
  i32x8 g1;
  g1[0] = (int)(1u << 16);                                   // data_size=1 (2B)
  g1[1] = (int)((dim0 & 0xFFFFu) << 16);                     // dim0[15:0]
  g1[2] = (int)(((dim0 >> 16) & 0xFFFFu) | ((dim1 & 0xFFFFu) << 16));
  g1[3] = (int)(((dim1 >> 16) & 0xFFFFu) | ((tile_k & 0xFFFFu) << 16));
  g1[4] = (int)(rows & 0xFFFFu);                             // tile_dim1
  g1[5] = (int)(unsigned)stride_elems;                       // dim0_stride lo
  g1[6] = (int)((unsigned)(stride_elems >> 32) & 0xFFFFu);   // dim0_stride hi
  g1[7] = 0;
  i32x4 gz; gz.x = 0; gz.y = 0; gz.z = 0; gz.w = 0;
#if defined(__clang_major__) && (__clang_major__ >= 23)
  i32x8 gz8;
  for (int i = 0; i < 8; i++) gz8[i] = 0;
  __builtin_amdgcn_tensor_load_to_lds(g0, g1, gz, gz, gz8, 0);
#else
  __builtin_amdgcn_tensor_load_to_lds(g0, g1, gz, gz, 0);
#endif
}

// grid-wide barrier: monotonic counter, target = phase * NWG
__device__ __forceinline__ void gsync(unsigned* bar, unsigned* phase) {
  __syncthreads();
  if (threadIdx.x == 0) {
    __threadfence();
    unsigned ph = ++(*phase);
    atomicAdd(bar, 1u);
    unsigned target = ph * NWG;
    while (__hip_atomic_load(bar, __ATOMIC_RELAXED, __HIP_MEMORY_SCOPE_AGENT) < target)
      __builtin_amdgcn_s_sleep(4);
    __threadfence();
  }
  __syncthreads();
}

// ---------- one LSTM layer step for 2 b-tiles (B frags reused 2x) ----------
__device__ __forceinline__ void lstm_layer2(
    const unsigned short* sA1, int rs1, int K1, const unsigned short* W1,
    const unsigned short* sA2, const unsigned short* W2, const float* bias,
    float* c, float* hf, unsigned short* hbf_out, int b_pair, int u_tile,
    int lane) {
  v8f acc[2][4];
#pragma unroll
  for (int bb = 0; bb < 2; bb++)
#pragma unroll
    for (int gg = 0; gg < 4; gg++) acc[bb][gg] = (v8f){};

  for (int k0 = 0; k0 < K1; k0 += 32) {
    int kt = k0 >> 5;
    v16bf w0 = load_b_frag(W1, kt, u_tile, 128, lane);
    v16bf w1 = load_b_frag(W1, kt, 32 + u_tile, 128, lane);
    v16bf w2 = load_b_frag(W1, kt, 64 + u_tile, 128, lane);
    v16bf w3 = load_b_frag(W1, kt, 96 + u_tile, 128, lane);
#pragma unroll
    for (int bb = 0; bb < 2; bb++) {
      v16bf a = load_a_lds(sA1, rs1, bb, k0, lane);
      acc[bb][0] = wmma_bf16(a, w0, acc[bb][0]);
      acc[bb][1] = wmma_bf16(a, w1, acc[bb][1]);
      acc[bb][2] = wmma_bf16(a, w2, acc[bb][2]);
      acc[bb][3] = wmma_bf16(a, w3, acc[bb][3]);
    }
  }
  for (int k0 = 0; k0 < Uq; k0 += 32) {
    int kt = k0 >> 5;
    v16bf w0 = load_b_frag(W2, kt, u_tile, 128, lane);
    v16bf w1 = load_b_frag(W2, kt, 32 + u_tile, 128, lane);
    v16bf w2 = load_b_frag(W2, kt, 64 + u_tile, 128, lane);
    v16bf w3 = load_b_frag(W2, kt, 96 + u_tile, 128, lane);
#pragma unroll
    for (int bb = 0; bb < 2; bb++) {
      v16bf a = load_a_lds(sA2, Uq, bb, k0, lane);
      acc[bb][0] = wmma_bf16(a, w0, acc[bb][0]);
      acc[bb][1] = wmma_bf16(a, w1, acc[bb][1]);
      acc[bb][2] = wmma_bf16(a, w2, acc[bb][2]);
      acc[bb][3] = wmma_bf16(a, w3, acc[bb][3]);
    }
  }
  int n = lane & 15, hk = lane >> 4;
  int col = u_tile * 16 + n;
  float bi = bias[col], bff = bias[512 + col], bg = bias[1024 + col],
        bo = bias[1536 + col];
#pragma unroll
  for (int bb = 0; bb < 2; bb++) {
#pragma unroll
    for (int r = 0; r < 8; r++) {
      int m = r + 8 * hk;
      int idx = ((b_pair * 2 + bb) * 16 + m) * Uq + col;
      float cn = fsig(acc[bb][1][r] + bff) * c[idx] +
                 fsig(acc[bb][0][r] + bi) * ftanh(acc[bb][2][r] + bg);
      float hn = fsig(acc[bb][3][r] + bo) * ftanh(cn);
      c[idx] = cn;
      hf[idx] = hn;
      hbf_out[idx] = f2bf(hn);
    }
  }
}

// ---------- params ----------
struct PParams {
  const unsigned short* xbf;
  const unsigned short *wx1, *wh1, *wx2, *wh2, *wx3, *wh3, *wout;
  const float *b1, *b2, *b3, *bout;
  unsigned short* h1b[2];
  unsigned short* h2b[2];
  unsigned short* h3b[2];
  float *c1, *c2, *c3, *h1f, *h2f, *h3f;
  float* out;
  unsigned* bar;
};

// ---------- persistent wavefront-pipelined LSTM ----------
__global__ __launch_bounds__(TPB) void lstm_persist(PParams P) {
  __shared__ unsigned short sA1[32 * Uq];  // layer-input tile (x or h_below)
  __shared__ unsigned short sA2[32 * Uq];  // recurrent h tile
  __shared__ float sm[16][Vq];
  __shared__ float rmax[16], rsum[16];
  __shared__ unsigned phase;
  int tid = threadIdx.x;
  int lane = tid & 31;
  int wv = tid >> 5;
  int w = blockIdx.x;
  // role: 0..31 -> layer1, 32..63 -> layer2, 64..95 -> layer3, 96..103 -> proj
  int layer = (w < 32) ? 1 : (w < 64) ? 2 : (w < 96) ? 3 : 0;
  int wl = (layer == 1) ? w : (layer == 2) ? w - 32 : (layer == 3) ? w - 64 : 0;
  int b_pair = wl >> 3;                // 0..3  (2 b-tiles each)
  int u_tile = (wl & 7) * 4 + wv;      // 0..31
  if (tid == 0) phase = 0;
  __syncthreads();

  float* probs = P.out;
  float* logits = P.out + (size_t)Bq * Sq * Vq;

  for (int d = 0; d < Sq + 3; d++) {
    if (layer == 1) {
      int t = d;
      if (t < Sq) {
        if (wv == 0) {
          tdm_load_2d(ldsoff(sA1),
                      P.xbf + (size_t)b_pair * 32 * (Sq * Vq) + (size_t)t * Vq,
                      Vq, 32, (unsigned long long)Sq * Vq);
          tdm_load_2d(ldsoff(sA2), P.h1b[t & 1] + (size_t)b_pair * 32 * Uq, Uq,
                      32, Uq);
          __builtin_amdgcn_s_wait_tensorcnt(0);
        }
        __syncthreads();
        lstm_layer2(sA1, Vq, Vq, P.wx1, sA2, P.wh1, P.b1, P.c1, P.h1f,
                    P.h1b[(t + 1) & 1], b_pair, u_tile, lane);
      }
    } else if (layer == 2) {
      int t = d - 1;
      if (t >= 0 && t < Sq) {
        if (wv == 0) {
          tdm_load_2d(ldsoff(sA1), P.h1b[(t + 1) & 1] + (size_t)b_pair * 32 * Uq,
                      Uq, 32, Uq);
          tdm_load_2d(ldsoff(sA2), P.h2b[t & 1] + (size_t)b_pair * 32 * Uq, Uq,
                      32, Uq);
          __builtin_amdgcn_s_wait_tensorcnt(0);
        }
        __syncthreads();
        lstm_layer2(sA1, Uq, Uq, P.wx2, sA2, P.wh2, P.b2, P.c2, P.h2f,
                    P.h2b[(t + 1) & 1], b_pair, u_tile, lane);
      }
    } else if (layer == 3) {
      int t = d - 2;
      if (t >= 0 && t < Sq) {
        if (wv == 0) {
          tdm_load_2d(ldsoff(sA1), P.h2b[(t + 1) & 1] + (size_t)b_pair * 32 * Uq,
                      Uq, 32, Uq);
          tdm_load_2d(ldsoff(sA2), P.h3b[t & 1] + (size_t)b_pair * 32 * Uq, Uq,
                      32, Uq);
          __builtin_amdgcn_s_wait_tensorcnt(0);
        }
        __syncthreads();
        lstm_layer2(sA1, Uq, Uq, P.wx3, sA2, P.wh3, P.b3, P.c3, P.h3f,
                    P.h3b[(t + 1) & 1], b_pair, u_tile, lane);
      }
    } else {
      // projection + softmax, one WG per b-tile, step t = d - 3
      int t = d - 3;
      if (t >= 0 && t < Sq) {
        int bt = w - 96;
        if (wv == 0) {
          tdm_load_2d(ldsoff(sA2), P.h3b[(t + 1) & 1] + (size_t)bt * 16 * Uq, Uq,
                      16, Uq);
          __builtin_amdgcn_s_wait_tensorcnt(0);
        }
        __syncthreads();
        v8f acc = {};
        for (int k0 = 0; k0 < Uq; k0 += 32) {
          v16bf a = load_a_lds(sA2, Uq, 0, k0, lane);
          acc = wmma_bf16(a, load_b_frag(P.wout, k0 >> 5, wv, 4, lane), acc);
        }
        int n = lane & 15, hk = lane >> 4;
        int col = wv * 16 + n;
        float bo = P.bout[col];
#pragma unroll
        for (int r = 0; r < 8; r++) {
          int m = r + 8 * hk;
          float lg = acc[r] + bo;
          sm[m][col] = lg;
          logits[(((size_t)(bt * 16 + m)) * Sq + t) * Vq + col] = lg;
        }
        __syncthreads();
        if (tid < 16) {
          float mx = -3.4e38f;
          for (int j = 0; j < Vq; j++) mx = fmaxf(mx, sm[tid][j]);
          float s = 0.f;
          for (int j = 0; j < Vq; j++) s += __expf(sm[tid][j] - mx);
          rmax[tid] = mx;
          rsum[tid] = s;
        }
        __syncthreads();
#pragma unroll
        for (int j = 0; j < 8; j++) {
          int e = tid * 8 + j;
          int m = e >> 6, cc = e & 63;
          probs[(((size_t)(bt * 16 + m)) * Sq + t) * Vq + cc] =
              __expf(sm[m][cc] - rmax[m]) / rsum[m];
        }
      }
    }
    gsync(P.bar, &phase);
  }

  // final carries h1,c1,h2,c2,h3,c3 -> d_out tail
  float* carry = P.out + 2 * (size_t)Bq * Sq * Vq;
  const float* srcs[6] = {P.h1f, P.c1, P.h2f, P.c2, P.h3f, P.c3};
  int gt = blockIdx.x * TPB + tid, nth = NWG * TPB;
  for (int s6 = 0; s6 < 6; s6++)
    for (int i = gt; i < Bq * Uq; i += nth)
      carry[(size_t)s6 * Bq * Uq + i] = srcs[s6][i];
}

// ---------- prep kernels ----------
__global__ void zero_ws(unsigned* p, int n) {
  int i = blockIdx.x * blockDim.x + threadIdx.x;
  if (i < n) p[i] = 0u;
}
__global__ void cvt_bf(const float* s, unsigned short* d, int n) {
  int i = blockIdx.x * blockDim.x + threadIdx.x;
  if (i < n) d[i] = f2bf(s[i]);
}
// pack fp32 [K][N] row-major -> bf16 WMMA B-fragment order
__global__ void pack_w(const float* W, unsigned short* o, int N, int total) {
  int idx = blockIdx.x * blockDim.x + threadIdx.x;
  if (idx >= total) return;
  int e = idx & 15;
  int lane = (idx >> 4) & 31;
  int frag = idx >> 9;
  int ntiles = N >> 4;
  int nt = frag % ntiles, kt = frag / ntiles;
  int n = nt * 16 + (lane & 15);
  int hk = lane >> 4;
  int kl = (e < 8) ? (hk * 8 + e) : (16 + hk * 8 + (e - 8));
  o[idx] = f2bf(W[(size_t)(kt * 32 + kl) * N + n]);
}

extern "C" void kernel_launch(void* const* d_in, const int* in_sizes, int n_in,
                              void* d_out, int out_size, void* d_ws,
                              size_t ws_size, hipStream_t stream) {
  (void)in_sizes; (void)n_in; (void)out_size; (void)ws_size;
  char* ws = (char*)d_ws;
  size_t off = 0;
  auto alloc = [&](size_t bytes) -> void* {
    void* p = ws + off;
    off = (off + bytes + 255) & ~(size_t)255;
    return p;
  };
  // zero-initialized region first (barrier + state buffers)
  unsigned* bar = (unsigned*)alloc(4);
  unsigned short* h1b0 = (unsigned short*)alloc(Bq * Uq * 2);
  unsigned short* h1b1 = (unsigned short*)alloc(Bq * Uq * 2);
  unsigned short* h2b0 = (unsigned short*)alloc(Bq * Uq * 2);
  unsigned short* h2b1 = (unsigned short*)alloc(Bq * Uq * 2);
  unsigned short* h3b0 = (unsigned short*)alloc(Bq * Uq * 2);
  unsigned short* h3b1 = (unsigned short*)alloc(Bq * Uq * 2);
  float* c1 = (float*)alloc(Bq * Uq * 4);
  float* c2 = (float*)alloc(Bq * Uq * 4);
  float* c3 = (float*)alloc(Bq * Uq * 4);
  float* h1f = (float*)alloc(Bq * Uq * 4);
  float* h2f = (float*)alloc(Bq * Uq * 4);
  float* h3f = (float*)alloc(Bq * Uq * 4);
  size_t zero_bytes = off;
  // converted / packed buffers
  unsigned short* xbf = (unsigned short*)alloc((size_t)Bq * Sq * Vq * 2);
  unsigned short* wx1p = (unsigned short*)alloc((size_t)Vq * 2048 * 2);
  unsigned short* wh1p = (unsigned short*)alloc((size_t)Uq * 2048 * 2);
  unsigned short* wx2p = (unsigned short*)alloc((size_t)Uq * 2048 * 2);
  unsigned short* wh2p = (unsigned short*)alloc((size_t)Uq * 2048 * 2);
  unsigned short* wx3p = (unsigned short*)alloc((size_t)Uq * 2048 * 2);
  unsigned short* wh3p = (unsigned short*)alloc((size_t)Uq * 2048 * 2);
  unsigned short* woutp = (unsigned short*)alloc((size_t)Uq * Vq * 2);

  int nz = (int)(zero_bytes / 4);
  zero_ws<<<(nz + 255) / 256, 256, 0, stream>>>((unsigned*)ws, nz);

  int nx = Bq * Sq * Vq;
  cvt_bf<<<(nx + 255) / 256, 256, 0, stream>>>((const float*)d_in[0], xbf, nx);

  int tw = Vq * 2048;
  pack_w<<<(tw + 255) / 256, 256, 0, stream>>>((const float*)d_in[1], wx1p, 2048, tw);
  tw = Uq * 2048;
  pack_w<<<(tw + 255) / 256, 256, 0, stream>>>((const float*)d_in[2], wh1p, 2048, tw);
  pack_w<<<(tw + 255) / 256, 256, 0, stream>>>((const float*)d_in[4], wx2p, 2048, tw);
  pack_w<<<(tw + 255) / 256, 256, 0, stream>>>((const float*)d_in[5], wh2p, 2048, tw);
  pack_w<<<(tw + 255) / 256, 256, 0, stream>>>((const float*)d_in[7], wx3p, 2048, tw);
  pack_w<<<(tw + 255) / 256, 256, 0, stream>>>((const float*)d_in[8], wh3p, 2048, tw);
  tw = Uq * Vq;
  pack_w<<<(tw + 255) / 256, 256, 0, stream>>>((const float*)d_in[10], woutp, Vq, tw);

  PParams P;
  P.xbf = xbf;
  P.wx1 = wx1p; P.wh1 = wh1p; P.wx2 = wx2p; P.wh2 = wh2p;
  P.wx3 = wx3p; P.wh3 = wh3p; P.wout = woutp;
  P.b1 = (const float*)d_in[3];
  P.b2 = (const float*)d_in[6];
  P.b3 = (const float*)d_in[9];
  P.bout = (const float*)d_in[11];
  P.h1b[0] = h1b0; P.h1b[1] = h1b1;
  P.h2b[0] = h2b0; P.h2b[1] = h2b1;
  P.h3b[0] = h3b0; P.h3b[1] = h3b1;
  P.c1 = c1; P.c2 = c2; P.c3 = c3;
  P.h1f = h1f; P.h2f = h2f; P.h3f = h3f;
  P.out = (float*)d_out;
  P.bar = bar;

  lstm_persist<<<NWG, TPB, 0, stream>>>(P);
}
